// VAttentionBlock_15152644620676
// MI455X (gfx1250) — matrix operand order repaired and checked by reading, not compiled, not run on hardware
//
#include <hip/hip_runtime.h>
#include <stdint.h>

#define BB 4
#define CC 512
#define NN 4096          // H*W
#define HEADS 4
#define DD 128           // head dim
#define GROUPS 32

typedef __attribute__((ext_vector_type(16))) __bf16 v16bf;
typedef __attribute__((ext_vector_type(8)))  float  v8f;
typedef int i32x4 __attribute__((__vector_size__(16)));

union Frag { v16bf v; uint32_t u[8]; };

__device__ __forceinline__ uint16_t f2bf(float f) {
  union { float f; uint32_t u; } x; x.f = f;
  uint32_t r = x.u + 0x7FFFu + ((x.u >> 16) & 1u);
  return (uint16_t)(r >> 16);
}

// ---- async global -> LDS staging (CDNA5 GLOBAL_LOAD_ASYNC_TO_LDS_B128, ASYNCcnt) ----
#if defined(__has_builtin)
#if __has_builtin(__builtin_amdgcn_global_load_async_to_lds_b128)
#define HAVE_ASYNC_LDS 1
#endif
#endif

#ifdef HAVE_ASYNC_LDS
#define ASYNC_CP(lp, gp)                                                              \
  __builtin_amdgcn_global_load_async_to_lds_b128((i32x4*)(gp), (i32x4*)(lp), 0, 0)
#define ASYNC_WAIT() asm volatile("s_wait_asynccnt 0x0" ::: "memory")
#else
#define ASYNC_CP(lp, gp) do { *(i32x4*)(lp) = *(const i32x4*)(gp); } while (0)
#define ASYNC_WAIT() do {} while (0)
#endif

// ---------------- GroupNorm statistics: one block per (b, group) ----------------
__global__ void gn_stats_kernel(const float* __restrict__ x,
                                float* __restrict__ mean, float* __restrict__ rstd) {
  const int bg = blockIdx.x;                       // 0..B*GROUPS-1
  const int GSZ = (CC / GROUPS) * NN;              // 16*4096 contiguous elements
  const float* p = x + (size_t)bg * GSZ;
  float s = 0.f, s2 = 0.f;
  for (int i = threadIdx.x; i < GSZ; i += blockDim.x) {
    float v = p[i]; s += v; s2 += v * v;
  }
  #pragma unroll
  for (int off = 16; off >= 1; off >>= 1) {
    s  += __shfl_xor(s,  off, 32);
    s2 += __shfl_xor(s2, off, 32);
  }
  __shared__ float ls[8], ls2[8];
  const int wid = threadIdx.x >> 5;
  if ((threadIdx.x & 31) == 0) { ls[wid] = s; ls2[wid] = s2; }
  __syncthreads();
  if (threadIdx.x == 0) {
    float ts = 0.f, ts2 = 0.f;
    for (int i = 0; i < 8; i++) { ts += ls[i]; ts2 += ls2[i]; }
    float inv = 1.f / (float)GSZ;
    float m = ts * inv;
    float var = ts2 * inv - m * m;
    mean[bg] = m;
    rstd[bg] = rsqrtf(var + 1e-5f);
  }
}

// ------------- GroupNorm apply + transpose to [b][n][c], fp32 -> bf16 -------------
__global__ void gn_apply_kernel(const float* __restrict__ x,
                                const float* __restrict__ gamma, const float* __restrict__ beta,
                                const float* __restrict__ mean, const float* __restrict__ rstd,
                                uint16_t* __restrict__ xn) {
  size_t idx = (size_t)blockIdx.x * blockDim.x + threadIdx.x;   // over B*C*N
  int n = (int)(idx % NN);
  size_t t = idx / NN;
  int c = (int)(t % CC);
  int b = (int)(t / CC);
  int g = b * GROUPS + c / (CC / GROUPS);
  float v = (x[idx] - mean[g]) * rstd[g] * gamma[c] + beta[c];
  xn[((size_t)b * NN + n) * CC + c] = f2bf(v);
}

// ---------------- weight conversion fp32 -> bf16 ----------------
__global__ void wconv_kernel(const float* __restrict__ qkv_w, const float* __restrict__ out_w,
                             uint16_t* __restrict__ wq, uint16_t* __restrict__ wo) {
  int idx = blockIdx.x * blockDim.x + threadIdx.x;
  if (idx < 3 * CC * CC) wq[idx] = f2bf(qkv_w[idx]);
  if (idx < CC * CC)     wo[idx] = f2bf(out_w[idx]);
}

// ---------------- QKV projection GEMM: 64x32 tile per wave, 8 waves/block ----------------
// A = W [3C][C] row-major bf16, B = xn [n][c] bf16.
// Writes Q,K to [bh][n][d] (bf16), V to [bh][d][n] (bf16). Q scaled by d^-0.5.
__global__ void __launch_bounds__(256)
qkv_gemm_kernel(const uint16_t* __restrict__ W, const uint16_t* __restrict__ X,
                const float* __restrict__ bias,
                uint16_t* __restrict__ qt, uint16_t* __restrict__ kt, uint16_t* __restrict__ vb) {
  const int lane = threadIdx.x & 31;
  const int wave = threadIdx.x >> 5;       // 8 waves: 2 along o, 4 along n
  const int half = lane >> 4, r16 = lane & 15;
  const int b = blockIdx.z;
  const int obase = blockIdx.x * 128 + (wave >> 2) * 64;
  const int nbase = blockIdx.y * 128 + (wave & 3) * 32;
  const uint16_t* Xb = X + (size_t)b * NN * CC;

  v8f acc[4][2] = {};
  for (int kc = 0; kc < CC; kc += 32) {
    Frag a[4], bf[2];
    #pragma unroll
    for (int i = 0; i < 4; i++) {
      const uint16_t* ap = W + (size_t)(obase + i * 16 + r16) * CC + kc + 16 * half;
      #pragma unroll
      for (int v = 0; v < 8; v++) {
        int kk = 2 * v;
        a[i].u[v] = *(const uint32_t*)(ap + kk);
      }
    }
    #pragma unroll
    for (int j = 0; j < 2; j++) {
      const uint16_t* bp = Xb + (size_t)(nbase + j * 16 + r16) * CC + kc + 16 * half;
      __builtin_prefetch(bp + 32, 0, 1);
      #pragma unroll
      for (int v = 0; v < 8; v++) bf[j].u[v] = *(const uint32_t*)(bp + 2 * v);
    }
    #pragma unroll
    for (int i = 0; i < 4; i++)
      #pragma unroll
      for (int j = 0; j < 2; j++)
        acc[i][j] = __builtin_amdgcn_wmma_f32_16x16x32_bf16(
            false, a[i].v, false, bf[j].v, (short)0, acc[i][j], false, false);
  }

  const float qscale = 0.08838834764831845f;   // 128^-0.5
  #pragma unroll
  for (int i = 0; i < 4; i++)
    #pragma unroll
    for (int j = 0; j < 2; j++)
      #pragma unroll
      for (int r = 0; r < 8; r++) {
        int o = obase + i * 16 + r + 8 * half;
        int n = nbase + j * 16 + r16;
        float val = acc[i][j][r] + bias[o];
        if (o < CC) {
          int h = o >> 7, d = o & 127;
          qt[(((size_t)(b * HEADS + h)) * NN + n) * DD + d] = f2bf(val * qscale);
        } else if (o < 2 * CC) {
          int oo = o - CC; int h = oo >> 7, d = oo & 127;
          kt[(((size_t)(b * HEADS + h)) * NN + n) * DD + d] = f2bf(val);
        } else {
          int oo = o - 2 * CC; int h = oo >> 7, d = oo & 127;
          vb[(((size_t)(b * HEADS + h)) * DD + d) * NN + n] = f2bf(val);
        }
      }
}

// ---------------- Flash attention with async-LDS staged K/V tiles ----------------
// Block = 4 waves = 64 query rows. K tile (32x128) and V tile (128x32) double-buffered
// in LDS, staged cooperatively; next tile's copies overlap current tile's WMMAs.
#define PADK 136   // ushorts per K-tile row  (272B: 16-lane frag reads hit distinct banks)
#define PADV 40    // ushorts per V-tile row  (80B: 20*d mod 64 distinct for d=0..15)

__global__ void __launch_bounds__(128)
attn_kernel(const uint16_t* __restrict__ qt, const uint16_t* __restrict__ kt,
            const uint16_t* __restrict__ vb, const float* __restrict__ x,
            uint16_t* __restrict__ ores) {
  const int tid = threadIdx.x;
  const int lane = tid & 31, wave = tid >> 5;
  const int half = lane >> 4, r16 = lane & 15;
  const int bh = blockIdx.y, b = bh >> 2, h = bh & 3;
  const int row0 = blockIdx.x * 64 + wave * 16;

  const uint16_t* Q = qt + (size_t)bh * NN * DD;
  const uint16_t* K = kt + (size_t)bh * NN * DD;
  const uint16_t* V = vb + (size_t)bh * DD * NN;

  __shared__ __align__(16) uint16_t ldsK[2][32 * PADK];
  __shared__ __align__(16) uint16_t ldsV[2][128 * PADV];
  __shared__ __align__(16) uint16_t ldsP[4][16 * 36];
  uint16_t* P = &ldsP[wave][0];

  auto stage = [&](int buf, int m) {
    // K tile: 32 rows x 128 ushorts = 512 16B chunks, 4 per thread
    #pragma unroll
    for (int c = 0; c < 4; c++) {
      int id = tid + c * 128;
      int row = id >> 4, col = (id & 15) * 8;
      ASYNC_CP(&ldsK[buf][row * PADK + col], K + (size_t)(m + row) * DD + col);
    }
    // V tile: 128 rows x 32 ushorts = 512 16B chunks, 4 per thread
    #pragma unroll
    for (int c = 0; c < 4; c++) {
      int id = tid + c * 128;
      int row = id >> 2, col = (id & 3) * 8;
      ASYNC_CP(&ldsV[buf][row * PADV + col], V + (size_t)row * NN + m + col);
    }
  };

  // Preload Q A-fragments for all 128 d (4 chunks of K=32)
  Frag qf[4];
  #pragma unroll
  for (int dc = 0; dc < 4; dc++) {
    const uint16_t* qp = Q + (size_t)(row0 + r16) * DD + dc * 32 + 16 * half;
    #pragma unroll
    for (int v = 0; v < 8; v++) {
      int kk = 2 * v;
      qf[dc].u[v] = *(const uint32_t*)(qp + kk);
    }
  }

  v8f oacc[8] = {};
  float rmax[8], rsum[8];
  #pragma unroll
  for (int r = 0; r < 8; r++) { rmax[r] = -3.0e38f; rsum[r] = 0.f; }

  stage(0, 0);

  int step = 0;
  for (int m = 0; m < NN; m += 32, step ^= 1) {
    ASYNC_WAIT();          // our async copies into buf[step] have landed
    __syncthreads();       // everyone's copies landed; peers done reading buf[step^1]
    if (m + 32 < NN) stage(step ^ 1, m + 32);   // overlap next stage with compute

    const uint16_t* Kt = &ldsK[step][0];
    const uint16_t* Vt = &ldsV[step][0];

    // S = Q . K^T for 32 key columns (two 16x16 tiles), fragments from LDS
    v8f s0 = {}, s1 = {};
    #pragma unroll
    for (int dc = 0; dc < 4; dc++) {
      Frag bk0, bk1;
      const uint16_t* kp0 = Kt + (size_t)r16 * PADK + dc * 32 + 16 * half;
      const uint16_t* kp1 = kp0 + 16 * PADK;
      #pragma unroll
      for (int v = 0; v < 8; v++) {
        bk0.u[v] = *(const uint32_t*)(kp0 + 2 * v);
        bk1.u[v] = *(const uint32_t*)(kp1 + 2 * v);
      }
      s0 = __builtin_amdgcn_wmma_f32_16x16x32_bf16(false, qf[dc].v, false, bk0.v, (short)0, s0, false, false);
      s1 = __builtin_amdgcn_wmma_f32_16x16x32_bf16(false, qf[dc].v, false, bk1.v, (short)0, s1, false, false);
    }

    // Online softmax: row r lives in acc element r, within one 16-lane half
    #pragma unroll
    for (int r = 0; r < 8; r++) {
      float v0 = s0[r], v1 = s1[r];
      float mx = fmaxf(v0, v1);
      mx = fmaxf(mx, __shfl_xor(mx, 1, 32));
      mx = fmaxf(mx, __shfl_xor(mx, 2, 32));
      mx = fmaxf(mx, __shfl_xor(mx, 4, 32));
      mx = fmaxf(mx, __shfl_xor(mx, 8, 32));
      float nm = fmaxf(rmax[r], mx);
      float corr = __expf(rmax[r] - nm);
      rmax[r] = nm;
      float p0 = __expf(v0 - nm), p1 = __expf(v1 - nm);
      float ps = p0 + p1;
      ps += __shfl_xor(ps, 1, 32);
      ps += __shfl_xor(ps, 2, 32);
      ps += __shfl_xor(ps, 4, 32);
      ps += __shfl_xor(ps, 8, 32);
      rsum[r] = rsum[r] * corr + ps;
      #pragma unroll
      for (int j = 0; j < 8; j++) oacc[j][r] *= corr;
      int row = r + 8 * half;
      P[row * 36 + r16]      = f2bf(p0);
      P[row * 36 + r16 + 16] = f2bf(p1);
    }
    asm volatile("s_wait_dscnt 0" ::: "memory");   // P store -> load within wave

    // P as one A-fragment (16x32), V B-fragments from LDS, O += P.V
    Frag pf;
    {
      const uint16_t* pp = P + r16 * 36 + 16 * half;
      #pragma unroll
      for (int v = 0; v < 8; v++) {
        int kk = 2 * v;
        pf.u[v] = *(const uint32_t*)(pp + kk);
      }
    }
    #pragma unroll
    for (int j = 0; j < 8; j++) {
      Frag vf;
      const uint16_t* vp = Vt + (size_t)(j * 16 + r16) * PADV + 16 * half;
      #pragma unroll
      for (int v = 0; v < 8; v++) vf.u[v] = *(const uint32_t*)(vp + 2 * v);
      oacc[j] = __builtin_amdgcn_wmma_f32_16x16x32_bf16(false, pf.v, false, vf.v, (short)0, oacc[j], false, false);
    }
  }

  // Normalize, add residual, store (o + x) as bf16 in [b][n][c]
  #pragma unroll
  for (int r = 0; r < 8; r++) {
    float inv = 1.f / rsum[r];
    int n = row0 + r + 8 * half;
    #pragma unroll
    for (int j = 0; j < 8; j++) {
      int c = h * DD + j * 16 + r16;
      float val = oacc[j][r] * inv + x[((size_t)b * CC + c) * NN + n];
      ores[((size_t)b * NN + n) * CC + c] = f2bf(val);
    }
  }
}

// ---------------- Output projection GEMM + bias (fp32 out), 64x32 per wave ----------------
__global__ void __launch_bounds__(256)
out_gemm_kernel(const uint16_t* __restrict__ W, const uint16_t* __restrict__ X,
                const float* __restrict__ bias, float* __restrict__ out) {
  const int lane = threadIdx.x & 31;
  const int wave = threadIdx.x >> 5;
  const int half = lane >> 4, r16 = lane & 15;
  const int b = blockIdx.z;
  const int obase = blockIdx.x * 128 + (wave >> 2) * 64;
  const int nbase = blockIdx.y * 128 + (wave & 3) * 32;
  const uint16_t* Xb = X + (size_t)b * NN * CC;

  v8f acc[4][2] = {};
  for (int kc = 0; kc < CC; kc += 32) {
    Frag a[4], bf[2];
    #pragma unroll
    for (int i = 0; i < 4; i++) {
      const uint16_t* ap = W + (size_t)(obase + i * 16 + r16) * CC + kc + 16 * half;
      #pragma unroll
      for (int v = 0; v < 8; v++) {
        int kk = 2 * v;
        a[i].u[v] = *(const uint32_t*)(ap + kk);
      }
    }
    #pragma unroll
    for (int j = 0; j < 2; j++) {
      const uint16_t* bp = Xb + (size_t)(nbase + j * 16 + r16) * CC + kc + 16 * half;
      __builtin_prefetch(bp + 32, 0, 1);
      #pragma unroll
      for (int v = 0; v < 8; v++) bf[j].u[v] = *(const uint32_t*)(bp + 2 * v);
    }
    #pragma unroll
    for (int i = 0; i < 4; i++)
      #pragma unroll
      for (int j = 0; j < 2; j++)
        acc[i][j] = __builtin_amdgcn_wmma_f32_16x16x32_bf16(
            false, a[i].v, false, bf[j].v, (short)0, acc[i][j], false, false);
  }

  #pragma unroll
  for (int i = 0; i < 4; i++)
    #pragma unroll
    for (int j = 0; j < 2; j++)
      #pragma unroll
      for (int r = 0; r < 8; r++) {
        int o = obase + i * 16 + r + 8 * half;
        int n = nbase + j * 16 + r16;
        out[((size_t)b * CC + o) * NN + n] = acc[i][j][r] + bias[o];
      }
}

extern "C" void kernel_launch(void* const* d_in, const int* in_sizes, int n_in,
                              void* d_out, int out_size, void* d_ws, size_t ws_size,
                              hipStream_t stream) {
  const float* x      = (const float*)d_in[0];
  const float* gamma  = (const float*)d_in[1];
  const float* beta   = (const float*)d_in[2];
  const float* qkv_w  = (const float*)d_in[3];
  const float* qkv_b  = (const float*)d_in[4];
  const float* out_w  = (const float*)d_in[5];
  const float* out_b  = (const float*)d_in[6];
  float* out = (float*)d_out;

  char* ws = (char*)d_ws;
  size_t off = 0;
  auto alloc = [&](size_t bytes) -> void* {
    void* p = ws + off;
    off = (off + bytes + 255) & ~(size_t)255;
    return p;
  };
  float*    mean = (float*)alloc((size_t)BB * GROUPS * sizeof(float));
  float*    rstd = (float*)alloc((size_t)BB * GROUPS * sizeof(float));
  uint16_t* wq   = (uint16_t*)alloc((size_t)3 * CC * CC * 2);
  uint16_t* wo   = (uint16_t*)alloc((size_t)CC * CC * 2);
  uint16_t* xn   = (uint16_t*)alloc((size_t)BB * NN * CC * 2);
  uint16_t* qt   = (uint16_t*)alloc((size_t)BB * HEADS * NN * DD * 2);
  uint16_t* kt   = (uint16_t*)alloc((size_t)BB * HEADS * NN * DD * 2);
  uint16_t* vbuf = (uint16_t*)alloc((size_t)BB * HEADS * DD * NN * 2);
  uint16_t* ores = (uint16_t*)alloc((size_t)BB * NN * CC * 2);

  gn_stats_kernel<<<BB * GROUPS, 256, 0, stream>>>(x, mean, rstd);
  gn_apply_kernel<<<(BB * CC * NN) / 256, 256, 0, stream>>>(x, gamma, beta, mean, rstd, xn);
  wconv_kernel<<<(3 * CC * CC) / 256, 256, 0, stream>>>(qkv_w, out_w, wq, wo);
  qkv_gemm_kernel<<<dim3((3 * CC) / 128, NN / 128, BB), 256, 0, stream>>>(wq, xn, qkv_b, qt, kt, vbuf);
  attn_kernel<<<dim3(NN / 64, BB * HEADS), 128, 0, stream>>>(qt, kt, vbuf, x, ores);
  out_gemm_kernel<<<dim3(CC / 128, NN / 128, BB), 256, 0, stream>>>(wo, ores, out_b, out);
}
